// CrossAttention_41540923686914
// MI455X (gfx1250) — matrix operand order repaired and checked by reading, not compiled, not run on hardware
//
#include <hip/hip_runtime.h>
#include <math.h>

typedef _Float16 v16h __attribute__((ext_vector_type(16)));
typedef _Float16 v8h  __attribute__((ext_vector_type(8)));
typedef _Float16 v2h  __attribute__((ext_vector_type(2)));
typedef float    v8f  __attribute__((ext_vector_type(8)));
typedef float    v4f  __attribute__((ext_vector_type(4)));

#define BATCH 8
#define CDIM  256
#define NDIM  4096
#define MDIM  4096
#define MCH   32            // m-chunk staged in LDS per iteration
#define NW    8             // waves per workgroup
#define ROWS  16            // query rows per wave
#define KSTRIDE (CDIM + 8)  // halves per row of kT   (528B: 16B aligned, 16-lane conflict-free)
#define VSTRIDE (MCH + 8)   // halves per row of vS   (80B)
#define PSTRIDE (MCH + 8)   // halves per row of pS
#define LOG2E 1.44269504088896f

#if __has_builtin(__builtin_amdgcn_exp2f)
#define EXP2F __builtin_amdgcn_exp2f
#else
#define EXP2F exp2f
#endif

__global__ __launch_bounds__(NW * 32)
void flash_xattn_f16(const float* __restrict__ h,
                     const float* __restrict__ xs,
                     const float* __restrict__ ys,
                     float* __restrict__ out) {
  // K chunk transposed [m][c], V chunk flat [c][m], P tiles per wave [n][m]
  __shared__ _Float16 kT[MCH][KSTRIDE];
  __shared__ _Float16 vS[CDIM][VSTRIDE];
  __shared__ _Float16 pS[NW][ROWS][PSTRIDE];

  const int tid  = threadIdx.x;
  const int w    = tid >> 5;
  const int lane = tid & 31;
  const int hi   = lane >> 4;   // which 16-lane half
  const int col  = lane & 15;   // D-column / A-row within fragments
  const int b    = blockIdx.y;
  const int n0   = blockIdx.x * (NW * ROWS) + w * ROWS;

  // ---- Load Q fragments once (f32 -> f16, log2e folded in). 8 chunks cover C=256.
  // A-layout (16x32 f16): i<8 -> k = hi*8+i ; i>=8 -> k = 16 + hi*8 + (i-8)
  v16h qa[8];
#pragma unroll
  for (int kc = 0; kc < 8; ++kc) {
#pragma unroll
    for (int i = 0; i < 16; ++i) {
      int k = (i < 8) ? (hi * 8 + i) : (16 + hi * 8 + (i - 8));
      int c = kc * 32 + k;
      float q = __builtin_nontemporal_load(&h[((size_t)(b * CDIM + c)) * NDIM + n0 + col]);
      qa[kc][i] = (_Float16)(q * LOG2E);
    }
  }

  // ---- flash accumulator: out tile rows n (r+8*hi), cols c (16*ct + col)
  v8f acc[16];
#pragma unroll
  for (int t = 0; t < 16; ++t)
#pragma unroll
    for (int j = 0; j < 8; ++j) acc[t][j] = 0.0f;

  float mrow[8], lrow[8];
#pragma unroll
  for (int r = 0; r < 8; ++r) { mrow[r] = -1e30f; lrow[r] = 0.0f; }

  for (int mb = 0; mb < MDIM / MCH; ++mb) {
    const size_t m0 = (size_t)mb * MCH;
    __syncthreads();  // previous iteration's readers done before restaging

    // ---- stage K chunk transposed: wave w converts channels [w*32, w*32+32)
    // two channels packed per store: v_cvt_pk_f16_f32 + ds_store_b32
    {
      const float* kp = xs + ((size_t)(b * CDIM + w * 32)) * MDIM + m0 + lane;
#pragma unroll
      for (int cc = 0; cc < 16; ++cc) {
        float x0 = __builtin_nontemporal_load(&kp[(size_t)(2 * cc) * MDIM]);
        float x1 = __builtin_nontemporal_load(&kp[(size_t)(2 * cc + 1) * MDIM]);
        v2h p; p[0] = (_Float16)x0; p[1] = (_Float16)x1;
        *(v2h*)&kT[lane][w * 32 + 2 * cc] = p;
      }
    }
    // ---- stage V chunk flat: thread tid owns channel tid (coalesced 16B loads)
    {
      const v4f* vp = (const v4f*)(ys + ((size_t)(b * CDIM + tid)) * MDIM + m0);
#pragma unroll
      for (int j = 0; j < 8; ++j) {
        v4f f = __builtin_nontemporal_load(&vp[j]);
        vS[tid][j * 4 + 0] = (_Float16)f[0];
        vS[tid][j * 4 + 1] = (_Float16)f[1];
        vS[tid][j * 4 + 2] = (_Float16)f[2];
        vS[tid][j * 4 + 3] = (_Float16)f[3];
      }
    }
    __syncthreads();

    // ---- S = Q K^T : two 16x16 logit tiles (m halves), contraction C=256 in 8 WMMAs each
    v8f s[2];
#pragma unroll
    for (int mt = 0; mt < 2; ++mt) {
      v8f st;
#pragma unroll
      for (int j = 0; j < 8; ++j) st[j] = 0.0f;
#pragma unroll
      for (int kc = 0; kc < 8; ++kc) {
        union { v16h v; v8h h2[2]; } bu;
        const v8h* bp = (const v8h*)&kT[mt * 16 + col][kc * 32 + hi * 8];
        bu.h2[0] = bp[0];
        bu.h2[1] = bp[2];
        st = __builtin_amdgcn_wmma_f32_16x16x32_f16(false, qa[kc], false, bu.v,
                                                    (short)0, st, false, false);
      }
      s[mt] = st;
    }

    // ---- online softmax (base-2; log2e already folded into Q)
    float mnew[8];
    int upd = 0;
#pragma unroll
    for (int r = 0; r < 8; ++r) {
      float mx = fmaxf(s[0][r], s[1][r]);
      mx = fmaxf(mx, __shfl_xor(mx, 1, 32));
      mx = fmaxf(mx, __shfl_xor(mx, 2, 32));
      mx = fmaxf(mx, __shfl_xor(mx, 4, 32));
      mx = fmaxf(mx, __shfl_xor(mx, 8, 32));
      mnew[r] = fmaxf(mrow[r], mx);
      upd |= (mnew[r] > mrow[r]) ? 1 : 0;
    }
    // wave-uniform skip: exp2(0)==1 exactly, so rescale is a no-op unless a max grew.
    if (__ballot(upd) != 0ull) {
#pragma unroll
      for (int r = 0; r < 8; ++r) {
        float alpha = EXP2F(mrow[r] - mnew[r]);
        mrow[r] = mnew[r];
        lrow[r] *= alpha;
#pragma unroll
        for (int t = 0; t < 16; ++t) acc[t][r] *= alpha;
      }
    }
#pragma unroll
    for (int r = 0; r < 8; ++r) {
      float p0 = EXP2F(s[0][r] - mrow[r]);
      float p1 = EXP2F(s[1][r] - mrow[r]);
      float rs = p0 + p1;
      rs += __shfl_xor(rs, 1, 32);
      rs += __shfl_xor(rs, 2, 32);
      rs += __shfl_xor(rs, 4, 32);
      rs += __shfl_xor(rs, 8, 32);
      lrow[r] += rs;
      // D-layout -> LDS: row n = r + 8*hi, cols m = col / 16+col
      pS[w][r + 8 * hi][col]      = (_Float16)p0;
      pS[w][r + 8 * hi][16 + col] = (_Float16)p1;
    }

    // ---- P (A-layout via own-wave LDS, DS ops are in-order per wave) x V tiles
    union { v16h v; v8h h2[2]; } pa;
    pa.h2[0] = *(const v8h*)&pS[w][col][hi * 8];
    pa.h2[1] = *(const v8h*)&pS[w][col][16 + hi * 8];
#pragma unroll
    for (int ct = 0; ct < 16; ++ct) {
      union { v16h v; v8h h2[2]; } bv;
      const v8h* vp2 = (const v8h*)&vS[ct * 16 + col][hi * 8];
      bv.h2[0] = vp2[0];
      bv.h2[1] = vp2[2];
      acc[ct] = __builtin_amdgcn_wmma_f32_16x16x32_f16(false, pa.v, false, bv.v,
                                                       (short)0, acc[ct], false, false);
    }
  }

  // ---- normalize and store out[b][c][n] (write-once: non-temporal)
  float inv[8];
#pragma unroll
  for (int r = 0; r < 8; ++r) inv[r] = 1.0f / lrow[r];
#pragma unroll
  for (int ct = 0; ct < 16; ++ct) {
    int c = ct * 16 + col;
    float* op = out + ((size_t)(b * CDIM + c)) * NDIM + n0;
#pragma unroll
    for (int r = 0; r < 8; ++r)
      __builtin_nontemporal_store(acc[ct][r] * inv[r], &op[r + 8 * hi]);
  }
}

extern "C" void kernel_launch(void* const* d_in, const int* in_sizes, int n_in,
                              void* d_out, int out_size, void* d_ws, size_t ws_size,
                              hipStream_t stream) {
  const float* h  = (const float*)d_in[0];
  const float* xs = (const float*)d_in[1];
  const float* ys = (const float*)d_in[2];
  float* out = (float*)d_out;
  (void)in_sizes; (void)n_in; (void)out_size; (void)d_ws; (void)ws_size;

  dim3 grid(NDIM / (NW * ROWS), BATCH);  // 32 x 8 workgroups
  flash_xattn_f16<<<grid, NW * 32, 0, stream>>>(h, xs, ys, out);
}
